// CrossMed4_85177791414322
// MI455X (gfx1250) — matrix operand order repaired and checked by reading, not compiled, not run on hardware
//
#include <hip/hip_runtime.h>
#include <hip/hip_bf16.h>
#include <math.h>

// ---------- problem constants ----------
#define BB 16
#define VV 16
#define MM 32
#define LL 24
#define DD 128
#define GG 384            // 3*D
#define OUTN 193
#define NM  (BB*VV)       // 256 monitor-GRU batch rows
#define NKM 5
#define NKV 7

typedef __attribute__((ext_vector_type(16))) _Float16 v16h;
typedef __attribute__((ext_vector_type(8)))  _Float16 v8h;
typedef __attribute__((ext_vector_type(8)))  float    v8f;

union AFrag { v16h v; v8h h[2]; };

// Load A fragment (16x32 f16, row-major, leading dim ld halves).
// ISA layout: lanes 0-15 -> M=lane, K={0..7,16..23}; lanes 16-31 -> M=lane-16, K={8..15,24..31}.
__device__ __forceinline__ v16h load_a_frag(const _Float16* __restrict__ base, int ld) {
    int lane = threadIdx.x & 31;
    int row  = lane & 15;
    int kb   = (lane >> 4) << 3;     // 0 or 8
    const _Float16* p = base + row * ld + kb;
    AFrag f;
    f.h[0] = *(const v8h*)(p);        // K = kb .. kb+7
    f.h[1] = *(const v8h*)(p + 16);   // K = kb+16 .. kb+23
    return f.v;
}

// Load B fragment for C = A * W^T where W is [N x K] row-major (so W rows are B columns).
// ISA layout: lane = column; lanes 0-15 hold K={0..7,16..23}, lanes 16-31 hold K={8..15,24..31}.
__device__ __forceinline__ v16h load_b_frag(const _Float16* __restrict__ W, int ld) {
    int lane = threadIdx.x & 31;
    int col  = lane & 15;
    int koff = (lane >> 4) << 3;     // 0 or 8
    const _Float16* p = W + col * ld + koff;
    AFrag f;
    f.h[0] = *(const v8h*)(p);
    f.h[1] = *(const v8h*)(p + 16);
    return f.v;
}

__device__ __forceinline__ float sigmoidf_(float x) { return 1.0f / (1.0f + __expf(-x)); }

// ---------- weight conversion ----------
__global__ __launch_bounds__(256) void f32_to_f16_kernel(const float* __restrict__ s,
                                                         _Float16* __restrict__ d, int n) {
    int i = blockIdx.x * 256 + threadIdx.x;
    if (i < n) d[i] = (_Float16)s[i];
}

// ---------- embedding kernels ----------
// e = sum_l emb[tok[n,l]]; broadcast across T monitor steps. grid=(NM), block=(DD)
__global__ __launch_bounds__(DD) void visit_embed_kernel(const int* __restrict__ tok,
                                                         const float* __restrict__ emb,
                                                         _Float16* __restrict__ monIn) {
    int n = blockIdx.x, d = threadIdx.x;
    const int* tk = tok + n * LL;
    float s = 0.f;
    #pragma unroll
    for (int l = 0; l < LL; ++l) s += emb[(size_t)tk[l] * DD + d];
    _Float16 hv = (_Float16)s;
    _Float16* o = monIn + (size_t)n * MM * DD + d;
    #pragma unroll
    for (int t = 0; t < MM; ++t) o[t * DD] = hv;
}

// lab/inj: sum_l embI[ti]*embV[tv].  grid=(NM*MM), block=(DD)
__global__ __launch_bounds__(DD) void pair_embed_kernel(const int* __restrict__ tokI,
                                                        const int* __restrict__ tokV,
                                                        const float* __restrict__ embI,
                                                        const float* __restrict__ embV,
                                                        _Float16* __restrict__ monIn) {
    int nm = blockIdx.x, d = threadIdx.x;
    const int* ti = tokI + nm * LL;
    const int* tv = tokV + nm * LL;
    float s = 0.f;
    #pragma unroll
    for (int l = 0; l < LL; ++l)
        s += embI[(size_t)ti[l] * DD + d] * embV[(size_t)tv[l] * DD + d];
    monIn[(size_t)nm * DD + d] = (_Float16)s;
}

// weight/age Linear(1,D). grid=(NM), block=(DD)
__global__ __launch_bounds__(DD) void info_embed_kernel(const float* __restrict__ val,
                                                        const float* __restrict__ iw,
                                                        const float* __restrict__ ib,
                                                        _Float16* __restrict__ visIn) {
    int n = blockIdx.x, d = threadIdx.x;
    visIn[(size_t)n * DD + d] = (_Float16)(val[n] * iw[d] + ib[d]);
}

// ---------- input-gate GEMM:  XG[r,g] = sum_k X[r,k]*W_key[g,k] + bih_key[g] ----------
// X: [rows, 128] f16 ; W: [nk, 384, 128] f16 ; XG: [rows, 384] f32
// 8 waves / block, 1 16x16 tile per wave; grid covers rowTiles*24 tiles exactly.
__global__ __launch_bounds__(256) void gemm_xg_kernel(const _Float16* __restrict__ X,
                                                      const _Float16* __restrict__ W,
                                                      const float* __restrict__ bih,
                                                      float* __restrict__ XG,
                                                      int rowsPerKeyT, int rowTilesTotal) {
    int wave = threadIdx.x >> 5;
    int lane = threadIdx.x & 31;
    int tileId = blockIdx.x * 8 + wave;
    int totalTiles = rowTilesTotal * (GG / 16);
    if (tileId >= totalTiles) return;
    int rowTile = tileId / (GG / 16);
    int nTile   = tileId % (GG / 16);
    int key     = (rowTile * 16) / rowsPerKeyT;

    const _Float16* Xt = X + (size_t)rowTile * 16 * DD;
    const _Float16* Wk = W + ((size_t)key * GG + (size_t)nTile * 16) * DD;
    const float*    bk = bih + (size_t)key * GG + nTile * 16;

    v16h a[4];
    #pragma unroll
    for (int kk = 0; kk < 4; ++kk) a[kk] = load_a_frag(Xt + kk * 32, DD);

    v8f acc = {};
    #pragma unroll
    for (int kk = 0; kk < 4; ++kk) {
        v16h b = load_b_frag(Wk + kk * 32, DD);
        acc = __builtin_amdgcn_wmma_f32_16x16x32_f16(false, a[kk], false, b,
                                                     (short)0, acc, false, false);
    }
    int col = nTile * 16 + (lane & 15);
    float bc = bk[lane & 15];
    int rbase = rowTile * 16 + ((lane >> 4) << 3);
    #pragma unroll
    for (int i = 0; i < 8; ++i)
        XG[(size_t)(rbase + i) * GG + col] = acc[i] + bc;
}

// ---------- GRU recurrence over a 16-row batch slice ----------
// Whh fragments are held in REGISTERS across the whole time loop (12 x v16h per wave),
// so each serial step only needs 4 ds_load A-fragments + 12 WMMAs + f32 gate math.
// Next step's xg rows are prefetched (global_prefetch_b8) during the gate phase.
__global__ __launch_bounds__(256) void gru_recur_kernel(const float* __restrict__ XG,
                                                        const _Float16* __restrict__ Whh,
                                                        const float* __restrict__ bhh,
                                                        int T, int rowsPerKey, int chunksPerKey,
                                                        _Float16* __restrict__ out16,
                                                        size_t out16KeyStride,
                                                        float* __restrict__ out32,
                                                        size_t out32KeyStride) {
    __shared__ float    h32[16 * DD];
    __shared__ _Float16 h16[16 * DD];
    __shared__ float    gh[16 * GG];

    int key   = blockIdx.x / chunksPerKey;
    int chunk = blockIdx.x % chunksPerKey;
    int nBase = chunk * 16;
    int tid = threadIdx.x, wave = tid >> 5, lane = tid & 31;

    for (int i = tid; i < 16 * DD; i += 256) { h32[i] = 0.f; h16[i] = (_Float16)0.f; }

    const _Float16* WhhK = Whh + (size_t)key * GG * DD;
    const float*    bhhK = bhh + (size_t)key * GG;
    const float*    xgK  = XG + (size_t)(key * rowsPerKey + nBase) * T * GG;

    // ---- preload this wave's 3 N-tiles of Whh into registers (stay for all T steps) ----
    v16h bw[3][4];
    float bc[3];
    #pragma unroll
    for (int tt = 0; tt < 3; ++tt) {
        int nTile = wave * 3 + tt;
        const _Float16* Wt = WhhK + (size_t)nTile * 16 * DD;
        #pragma unroll
        for (int kk = 0; kk < 4; ++kk) bw[tt][kk] = load_b_frag(Wt + kk * 32, DD);
        bc[tt] = bhhK[nTile * 16 + (lane & 15)];
    }
    __syncthreads();

    for (int t = 0; t < T; ++t) {
        v16h a[4];
        #pragma unroll
        for (int kk = 0; kk < 4; ++kk) a[kk] = load_a_frag(h16 + kk * 32, DD);

        #pragma unroll
        for (int tt = 0; tt < 3; ++tt) {
            int nTile = wave * 3 + tt;
            v8f acc = {};
            #pragma unroll
            for (int kk = 0; kk < 4; ++kk)
                acc = __builtin_amdgcn_wmma_f32_16x16x32_f16(false, a[kk], false, bw[tt][kk],
                                                             (short)0, acc, false, false);
            int col = nTile * 16 + (lane & 15);
            #pragma unroll
            for (int i = 0; i < 8; ++i) {
                int row = i + ((lane >> 4) << 3);
                gh[row * GG + col] = acc[i] + bc[tt];
            }
        }
        __syncthreads();

        for (int i = tid; i < 16 * DD; i += 256) {
            int n = i >> 7, d = i & 127;
            const float* xr = xgK + ((size_t)n * T + t) * GG;
            if (t + 1 < T) __builtin_prefetch(xr + GG + d, 0, 1);   // next step's gates
            float ir = xr[d], iz = xr[d + DD], in_ = xr[d + 2 * DD];
            float hr = gh[n * GG + d], hz = gh[n * GG + d + DD], hn = gh[n * GG + d + 2 * DD];
            float rg = sigmoidf_(ir + hr);
            float zg = sigmoidf_(iz + hz);
            float ng = tanhf(in_ + rg * hn);
            float hv = (1.f - zg) * ng + zg * h32[i];
            h32[i] = hv;
            h16[i] = (_Float16)hv;
        }
        __syncthreads();
    }

    for (int i = tid; i < 16 * DD; i += 256) {
        int n = i >> 7, d = i & 127;
        int gn = nBase + n;
        if (out16) out16[(size_t)key * out16KeyStride + (size_t)gn * DD + d] = h16[i];
        if (out32) out32[(size_t)key * out32KeyStride + (size_t)gn * DD + d] = h32[i];
    }
}

// ---------- final ReLU + Linear(896, 193), scalar f32 (tiny, precision-sensitive) ----------
__global__ __launch_bounds__(256) void fc_kernel(const float* __restrict__ visH,
                                                 const float* __restrict__ W,
                                                 const float* __restrict__ bias,
                                                 float* __restrict__ out) {
    int idx = blockIdx.x * 256 + threadIdx.x;
    if (idx >= BB * OUTN) return;
    int b = idx / OUTN, o = idx % OUTN;
    float s = bias[o];
    for (int j = 0; j < NKV * DD; ++j) {
        int k = j >> 7, d = j & 127;
        float x = visH[((size_t)k * BB + b) * DD + d];
        x = x > 0.f ? x : 0.f;
        s += x * W[(size_t)j * OUTN + o];
    }
    out[(size_t)b * OUTN + o] = s;
}

extern "C" void kernel_launch(void* const* d_in, const int* in_sizes, int n_in,
                              void* d_out, int out_size, void* d_ws, size_t ws_size,
                              hipStream_t stream) {
    (void)in_sizes; (void)n_in; (void)out_size; (void)ws_size;

    const int*   tok_cond      = (const int*)d_in[0];
    const int*   tok_proc      = (const int*)d_in[1];
    const int*   tok_drug      = (const int*)d_in[2];
    const int*   tok_lab_item  = (const int*)d_in[3];
    const int*   tok_lab_value = (const int*)d_in[4];
    const int*   tok_inj_item  = (const int*)d_in[5];
    const int*   tok_inj_value = (const int*)d_in[6];
    const float* weight        = (const float*)d_in[7];
    const float* age           = (const float*)d_in[8];
    const float* emb_cond      = (const float*)d_in[9];
    const float* emb_proc      = (const float*)d_in[10];
    const float* emb_drug      = (const float*)d_in[11];
    const float* emb_lab_item  = (const float*)d_in[12];
    const float* emb_lab_value = (const float*)d_in[13];
    const float* emb_inj_item  = (const float*)d_in[14];
    const float* emb_inj_value = (const float*)d_in[15];
    const float* mgru_wih      = (const float*)d_in[16];
    const float* mgru_whh      = (const float*)d_in[17];
    const float* mgru_bih      = (const float*)d_in[18];
    const float* mgru_bhh      = (const float*)d_in[19];
    const float* vgru_wih      = (const float*)d_in[20];
    const float* vgru_whh      = (const float*)d_in[21];
    const float* vgru_bih      = (const float*)d_in[22];
    const float* vgru_bhh      = (const float*)d_in[23];
    const float* info_w        = (const float*)d_in[24];
    const float* info_b        = (const float*)d_in[25];
    const float* fc_w          = (const float*)d_in[26];
    const float* fc_b          = (const float*)d_in[27];
    float* out = (float*)d_out;

    // ---- workspace partition ----
    char* ws = (char*)d_ws;
    size_t off = 0;
    auto take = [&](size_t bytes) -> char* {
        char* p = ws + off;
        off = (off + bytes + 255) & ~(size_t)255;
        return p;
    };
    _Float16* monIn  = (_Float16*)take((size_t)NKM * NM * MM * DD * sizeof(_Float16)); // [5][256][32][128]
    float*    xg     = (float*)   take((size_t)NKM * NM * MM * GG * sizeof(float));    // [5][256][32][384]
    _Float16* visIn  = (_Float16*)take((size_t)NKV * NM * DD * sizeof(_Float16));      // [7][256][128]
    float*    visXg  = (float*)   take((size_t)NKV * NM * GG * sizeof(float));         // [7][256][384]
    float*    visH   = (float*)   take((size_t)NKV * BB * DD * sizeof(float));         // [7][16][128]
    _Float16* wihM16 = (_Float16*)take((size_t)NKM * GG * DD * sizeof(_Float16));
    _Float16* whhM16 = (_Float16*)take((size_t)NKM * GG * DD * sizeof(_Float16));
    _Float16* wihV16 = (_Float16*)take((size_t)NKV * GG * DD * sizeof(_Float16));
    _Float16* whhV16 = (_Float16*)take((size_t)NKV * GG * DD * sizeof(_Float16));

    // ---- 1. convert GRU weights to f16 ----
    {
        int nM = NKM * GG * DD, nV = NKV * GG * DD;
        f32_to_f16_kernel<<<(nM + 255) / 256, 256, 0, stream>>>(mgru_wih, wihM16, nM);
        f32_to_f16_kernel<<<(nM + 255) / 256, 256, 0, stream>>>(mgru_whh, whhM16, nM);
        f32_to_f16_kernel<<<(nV + 255) / 256, 256, 0, stream>>>(vgru_wih, wihV16, nV);
        f32_to_f16_kernel<<<(nV + 255) / 256, 256, 0, stream>>>(vgru_whh, whhV16, nV);
    }

    // ---- 2. embeddings -> monitor GRU input (f16) ----
    const size_t keyStrideMon = (size_t)NM * MM * DD;
    visit_embed_kernel<<<NM, DD, 0, stream>>>(tok_cond, emb_cond, monIn + 0 * keyStrideMon);
    visit_embed_kernel<<<NM, DD, 0, stream>>>(tok_proc, emb_proc, monIn + 1 * keyStrideMon);
    visit_embed_kernel<<<NM, DD, 0, stream>>>(tok_drug, emb_drug, monIn + 2 * keyStrideMon);
    pair_embed_kernel<<<NM * MM, DD, 0, stream>>>(tok_lab_item, tok_lab_value,
                                                  emb_lab_item, emb_lab_value,
                                                  monIn + 3 * keyStrideMon);
    pair_embed_kernel<<<NM * MM, DD, 0, stream>>>(tok_inj_item, tok_inj_value,
                                                  emb_inj_item, emb_inj_value,
                                                  monIn + 4 * keyStrideMon);

    // ---- 3. patient info -> visit GRU input keys 5,6 (f16) ----
    info_embed_kernel<<<NM, DD, 0, stream>>>(weight, info_w + 0 * DD, info_b + 0 * DD,
                                             visIn + (size_t)5 * NM * DD);
    info_embed_kernel<<<NM, DD, 0, stream>>>(age, info_w + 1 * DD, info_b + 1 * DD,
                                             visIn + (size_t)6 * NM * DD);

    // ---- 4. monitor input-gate GEMM (WMMA): [40960,128]x[128,384] ----
    {
        int rowTiles = (NKM * NM * MM) / 16;           // 2560
        int blocks   = (rowTiles * (GG / 16) + 7) / 8; // 7680
        gemm_xg_kernel<<<blocks, 256, 0, stream>>>(monIn, wihM16, mgru_bih, xg,
                                                   NM * MM, rowTiles);
    }

    // ---- 5. monitor GRU recurrence (WMMA per step), final h -> visIn keys 0..4 ----
    gru_recur_kernel<<<NKM * (NM / 16), 256, 0, stream>>>(xg, whhM16, mgru_bhh,
                                                          MM, NM, NM / 16,
                                                          visIn, (size_t)NM * DD,
                                                          nullptr, 0);

    // ---- 6. visit input-gate GEMM (WMMA): [1792,128]x[128,384] ----
    {
        int rowTiles = (NKV * NM) / 16;                // 112
        int blocks   = (rowTiles * (GG / 16) + 7) / 8; // 336
        gemm_xg_kernel<<<blocks, 256, 0, stream>>>(visIn, wihV16, vgru_bih, visXg,
                                                   NM, rowTiles);
    }

    // ---- 7. visit GRU recurrence (batch=B=16 rows per key, T=V=16) ----
    gru_recur_kernel<<<NKV, 256, 0, stream>>>(visXg, whhV16, vgru_bhh,
                                              VV, BB, 1,
                                              nullptr, 0,
                                              visH, (size_t)BB * DD);

    // ---- 8. ReLU + final Linear ----
    fc_kernel<<<(BB * OUTN + 255) / 256, 256, 0, stream>>>(visH, fc_w, fc_b, out);
}
